// FastenHEAT_21955872817583
// MI455X (gfx1250) — matrix operand order repaired and checked by reading, not compiled, not run on hardware
//
#include <hip/hip_runtime.h>
#include <hip/hip_bf16.h>

#define GNT 3
#define GETE 5
#define GEAE 6
#define SLOPE 0.2f

typedef __attribute__((ext_vector_type(16))) _Float16 v16h;
typedef __attribute__((ext_vector_type(8)))  float    v8f;
typedef int v4i __attribute__((vector_size(16)));

#if __has_builtin(__builtin_amdgcn_global_load_async_to_lds_b128)
#define ASYNC_LDS 1
#endif

__device__ __forceinline__ float leaky(float v) { return v > 0.0f ? v : SLOPE * v; }

// ---- monotonic float<->uint key for atomic max over floats ----
__device__ __forceinline__ unsigned fkey(float v) {
    unsigned u = __float_as_uint(v);
    return (u & 0x80000000u) ? ~u : (u | 0x80000000u);
}
__device__ __forceinline__ float fdecode(unsigned k) {
    unsigned u = (k & 0x80000000u) ? (k ^ 0x80000000u) : ~k;
    return __uint_as_float(u);
}

// ---- async global->LDS copy of one 16B chunk (CDNA5 path, safe fallback) ----
__device__ __forceinline__ void async_copy_b128(const _Float16* __restrict__ g,
                                                _Float16* l) {
#ifdef ASYNC_LDS
    __builtin_amdgcn_global_load_async_to_lds_b128(
        (__attribute__((address_space(1))) v4i*)g,
        (__attribute__((address_space(3))) v4i*)l, 0, 0);
#else
    *(v4i*)l = *(const v4i*)g;
#endif
}
__device__ __forceinline__ void async_join() {
#ifdef ASYNC_LDS
#if __has_builtin(__builtin_amdgcn_s_wait_asynccnt)
    __builtin_amdgcn_s_wait_asynccnt(0);
#else
    asm volatile("s_wait_asynccnt 0" ::: "memory");
#endif
#endif
    __syncthreads();
}

// ---- weight pre-conversion: f32 row-major [64][NC] -> f16 col-major [NC][64] ----
__global__ void cvt_colmajor_f16(const float* __restrict__ B, _Float16* __restrict__ o,
                                 int NC) {
    const int i = blockIdx.x * blockDim.x + threadIdx.x;
    if (i >= 64 * NC) return;
    const int col = i / 64, k = i % 64;
    o[col * 64 + k] = (_Float16)B[k * NC + col];
}

// ---- WMMA A fragment (f32 activations -> f16 fragment) ----
// lane<16: row=lane, elems 0..15 = K 0..15.
// lane>=16: row=lane-16, elems 0..15 = K 16..31.
__device__ __forceinline__ v16h a_frag(const float* __restrict__ A, int rowBase,
                                       int kchunk, int lane) {
    const int r    = lane & 15;
    const int koff = (lane >> 4) << 4;
    const float* p = A + (size_t)(rowBase + r) * 64 + kchunk + koff;
    v16h a;
#pragma unroll
    for (int e = 0; e < 8; ++e) a[e] = (_Float16)p[e];
#pragma unroll
    for (int e = 0; e < 8; ++e) a[8 + e] = (_Float16)p[8 + e];
    return a;
}
// ---- WMMA B fragment straight out of LDS (col-major f16: contiguous 32B) ----
__device__ __forceinline__ v16h b_frag_lds(const _Float16* sB, int kchunk, int colBase,
                                           int lane) {
    const int col = colBase + (lane & 15);
    const int kb  = kchunk + ((lane >> 4) << 4);
    return *(const v16h*)(sB + col * 64 + kb);
}

// ---- generic C = act(A[M,64] @ B[64,NCOLS] + bias), B16 col-major f16 ----
template <int NCOLS, bool RELU, bool BIAS>
__global__ void gemm64_wmma(const float* __restrict__ A, const _Float16* __restrict__ B16,
                            const float* __restrict__ bias, float* __restrict__ C, int M) {
    constexpr int HALVES = NCOLS * 64;
    __shared__ alignas(32) _Float16 sB[HALVES];
    for (int i = threadIdx.x; i < HALVES / 8; i += blockDim.x)
        async_copy_b128(B16 + i * 8, sB + i * 8);
    async_join();

    const int lane = threadIdx.x & 31;
    const int tile = blockIdx.x * (blockDim.x >> 5) + (threadIdx.x >> 5);
    const int rowBase = tile * 16;
    if (rowBase >= M) return; // wave-uniform after barrier: EXEC all-1s for WMMA

    const v16h a0 = a_frag(A, rowBase, 0, lane);
    const v16h a1 = a_frag(A, rowBase, 32, lane);
    const int col0 = lane & 15;
    const int mb   = (lane >> 4) << 3;

#pragma unroll
    for (int ct = 0; ct < NCOLS / 16; ++ct) {
        v8f c = {};
        c = __builtin_amdgcn_wmma_f32_16x16x32_f16(false, a0, false,
                b_frag_lds(sB, 0, ct * 16, lane), (short)0, c, false, false);
        c = __builtin_amdgcn_wmma_f32_16x16x32_f16(false, a1, false,
                b_frag_lds(sB, 32, ct * 16, lane), (short)0, c, false, false);
        const int col = ct * 16 + col0;
        const float bv = BIAS ? bias[col] : 0.0f;
#pragma unroll
        for (int r = 0; r < 8; ++r) {
            float v = c[r] + bv;
            if (RELU) v = v > 0.0f ? v : 0.0f;
            C[(size_t)(rowBase + mb + r) * NCOLS + col] = v;
        }
    }
}

// ---- HeteroLinear: all NT projections via WMMA (weights in LDS), per-row select ----
__global__ void hetero_gemm64(const float* __restrict__ A,
                              const _Float16* __restrict__ W16 /*[NT][64*64] col-major*/,
                              const float* __restrict__ bias /*[NT][64]*/,
                              const int* __restrict__ ntype, float* __restrict__ C, int M) {
    __shared__ alignas(32) _Float16 sB[GNT * 64 * 64];
    for (int i = threadIdx.x; i < GNT * 64 * 64 / 8; i += blockDim.x)
        async_copy_b128(W16 + i * 8, sB + i * 8);
    async_join();

    const int lane = threadIdx.x & 31;
    const int tile = blockIdx.x * (blockDim.x >> 5) + (threadIdx.x >> 5);
    const int rowBase = tile * 16;
    if (rowBase >= M) return;

    const v16h a0 = a_frag(A, rowBase, 0, lane);
    const v16h a1 = a_frag(A, rowBase, 32, lane);
    const int col0 = lane & 15;
    const int mb   = (lane >> 4) << 3;
    int nt[8];
#pragma unroll
    for (int r = 0; r < 8; ++r) nt[r] = ntype[rowBase + mb + r];

#pragma unroll
    for (int ct = 0; ct < 4; ++ct) {
        const int col = ct * 16 + col0;
        float outv[8];
#pragma unroll
        for (int r = 0; r < 8; ++r) outv[r] = 0.0f;
        for (int t = 0; t < GNT; ++t) {
            const _Float16* sBt = sB + t * 64 * 64;
            v8f c = {};
            c = __builtin_amdgcn_wmma_f32_16x16x32_f16(false, a0, false,
                    b_frag_lds(sBt, 0, ct * 16, lane), (short)0, c, false, false);
            c = __builtin_amdgcn_wmma_f32_16x16x32_f16(false, a1, false,
                    b_frag_lds(sBt, 32, ct * 16, lane), (short)0, c, false, false);
            const float bv = bias[t * 64 + col];
#pragma unroll
            for (int r = 0; r < 8; ++r)
                if (nt[r] == t) outv[r] = c[r] + bv;
        }
#pragma unroll
        for (int r = 0; r < 8; ++r)
            C[(size_t)(rowBase + mb + r) * 64 + col] = outv[r];
    }
}

// ---- per-node attention projections: ad = hw@attW[0:64], as = hw@attW[64:128] ----
__global__ void node_att_proj(const float* __restrict__ hw,
                              const float* __restrict__ attW /*[139][2]*/,
                              float* __restrict__ ad, float* __restrict__ as_, int Nn) {
    __shared__ float sW[256];
    for (int i = threadIdx.x; i < 256; i += blockDim.x) sW[i] = attW[i];
    __syncthreads();
    const int n = blockIdx.x * blockDim.x + threadIdx.x;
    if (n >= Nn) return;
    const float* hrow = hw + (size_t)n * 64;
    float d0 = 0, d1 = 0, s0 = 0, s1 = 0;
#pragma unroll 8
    for (int k = 0; k < 64; ++k) {
        const float hv = hrow[k];
        d0 += hv * sW[k * 2 + 0];
        d1 += hv * sW[k * 2 + 1];
        s0 += hv * sW[(64 + k) * 2 + 0];
        s1 += hv * sW[(64 + k) * 2 + 1];
    }
    ad[n * 2] = d0;  ad[n * 2 + 1] = d1;
    as_[n * 2] = s0; as_[n * 2 + 1] = s1;
}

// ---- edge pass 1: logits + segment max (monotonic uint key) ----
__global__ void edge_logits(const int* __restrict__ ei, const int* __restrict__ etype,
                            const float* __restrict__ eattr,
                            const float* __restrict__ ad, const float* __restrict__ as_,
                            const float* __restrict__ etemb /*[ET][5]*/,
                            const float* __restrict__ eaW /*[2][6]*/,
                            const float* __restrict__ attW /*[139][2]*/,
                            float* __restrict__ alpha, unsigned* __restrict__ mkey, int Ne) {
    const int e = blockIdx.x * blockDim.x + threadIdx.x;
    if (e >= Ne) return;
    const int s = ei[e], d = ei[Ne + e];
    const int t = etype[e];
    const float ea0 = eattr[2 * e], ea1 = eattr[2 * e + 1];
    float l0 = ad[2 * d]     + as_[2 * s];
    float l1 = ad[2 * d + 1] + as_[2 * s + 1];
#pragma unroll
    for (int j = 0; j < GETE; ++j) {
        const float v = leaky(etemb[t * GETE + j]);
        l0 += v * attW[(128 + j) * 2 + 0];
        l1 += v * attW[(128 + j) * 2 + 1];
    }
#pragma unroll
    for (int j = 0; j < GEAE; ++j) {
        const float v = leaky(ea0 * eaW[j] + ea1 * eaW[GEAE + j]);
        l0 += v * attW[(133 + j) * 2 + 0];
        l1 += v * attW[(133 + j) * 2 + 1];
    }
    l0 = leaky(l0); l1 = leaky(l1);
    alpha[2 * e] = l0; alpha[2 * e + 1] = l1;
    atomicMax(&mkey[2 * d],     fkey(l0));
    atomicMax(&mkey[2 * d + 1], fkey(l1));
}

// ---- edge pass 2: a = exp(alpha - m[dst]); segment sum ----
__global__ void edge_exp(const int* __restrict__ ei, const unsigned* __restrict__ mkey,
                         float* __restrict__ alpha, float* __restrict__ ssum, int Ne) {
    const int e = blockIdx.x * blockDim.x + threadIdx.x;
    if (e >= Ne) return;
    const int d = ei[Ne + e];
    const float a0 = __expf(alpha[2 * e]     - fdecode(mkey[2 * d]));
    const float a1 = __expf(alpha[2 * e + 1] - fdecode(mkey[2 * d + 1]));
    alpha[2 * e] = a0; alpha[2 * e + 1] = a1;
    atomicAdd(&ssum[2 * d], a0);
    atomicAdd(&ssum[2 * d + 1], a1);
}

// ---- edge pass 3: acc[dst] += w * (msgn[src] + eae @ linW2); 4 threads/edge ----
__global__ void edge_scatter(const int* __restrict__ ei, const float* __restrict__ eattr,
                             const float* __restrict__ eaW /*[2][6]*/,
                             const float* __restrict__ linW2 /*[6][64]*/,
                             const float* __restrict__ msgn,
                             const float* __restrict__ aexp, const float* __restrict__ ssum,
                             float* __restrict__ acc, int Ne) {
    __shared__ float sW2[GEAE * 64];
    __shared__ float sEa[2 * GEAE];
    for (int i = threadIdx.x; i < GEAE * 64; i += blockDim.x) sW2[i] = linW2[i];
    if (threadIdx.x < 2 * GEAE) sEa[threadIdx.x] = eaW[threadIdx.x];
    __syncthreads();

    const int tid = blockIdx.x * blockDim.x + threadIdx.x;
    const int e = tid >> 2;
    if (e >= Ne) return;
    const int c = (tid & 3) * 16;
    const int s = ei[e], d = ei[Ne + e];
    const float w = 0.5f * (aexp[2 * e]     / (ssum[2 * d]     + 1e-16f) +
                            aexp[2 * e + 1] / (ssum[2 * d + 1] + 1e-16f));
    const float ea0 = eattr[2 * e], ea1 = eattr[2 * e + 1];
    float eae[GEAE];
#pragma unroll
    for (int j = 0; j < GEAE; ++j) eae[j] = leaky(ea0 * sEa[j] + ea1 * sEa[GEAE + j]);
    const float* ms = msgn + (size_t)s * 64 + c;
    float* ap = acc + (size_t)d * 64 + c;
#pragma unroll
    for (int i = 0; i < 16; ++i) {
        float v = ms[i];
#pragma unroll
        for (int j = 0; j < GEAE; ++j) v += eae[j] * sW2[j * 64 + c + i];
        atomicAdd(&ap[i], w * v);
    }
}

extern "C" void kernel_launch(void* const* d_in, const int* in_sizes, int n_in,
                              void* d_out, int out_size, void* d_ws, size_t ws_size,
                              hipStream_t stream) {
    const float* x         = (const float*)d_in[0];
    const int*   edge_idx  = (const int*)d_in[1];
    const int*   node_type = (const int*)d_in[2];
    const int*   edge_type = (const int*)d_in[3];
    const float* edge_attr = (const float*)d_in[4];
    const float* W_in      = (const float*)d_in[5];
    const float* b_in      = (const float*)d_in[6];
    const float* hl_W      = (const float*)d_in[7];
    const float* hl_b      = (const float*)d_in[8];
    const float* et_emb    = (const float*)d_in[9];
    const float* ea_W      = (const float*)d_in[10];
    const float* att_W     = (const float*)d_in[11];
    const float* lin_W     = (const float*)d_in[12];
    const float* W_out     = (const float*)d_in[13];
    const float* b_out     = (const float*)d_in[14];
    float* out = (float*)d_out;

    const int N = in_sizes[2];
    const int E = in_sizes[3];

    // workspace carve-up (acc aliases h: h is dead after HeteroLinear)
    char* w = (char*)d_ws;
    float* h     = (float*)w; w += (size_t)N * 64 * sizeof(float);
    float* hw    = (float*)w; w += (size_t)N * 64 * sizeof(float);
    float* msgn  = (float*)w; w += (size_t)N * 64 * sizeof(float);
    float* ad    = (float*)w; w += (size_t)N * 2 * sizeof(float);
    float* as_   = (float*)w; w += (size_t)N * 2 * sizeof(float);
    unsigned* mk = (unsigned*)w; w += (size_t)N * 2 * sizeof(unsigned);
    float* ssum  = (float*)w; w += (size_t)N * 2 * sizeof(float);
    float* alpha = (float*)w; w += (size_t)E * 2 * sizeof(float);
    _Float16* w16_in  = (_Float16*)w; w += 64 * 64 * sizeof(_Float16);
    _Float16* w16_hl  = (_Float16*)w; w += 2 * GNT * 64 * 64 * sizeof(_Float16);
    _Float16* w16_msg = (_Float16*)w; w += 2 * 64 * 64 * sizeof(_Float16);
    _Float16* w16_out = (_Float16*)w; w += 64 * 16 * sizeof(_Float16);

    const int tilesN   = N / 16;  // 6250, exact
    const int gemmGrid = (tilesN + 7) / 8;  // 8 waves (256 thr) per block
    const int nGrid    = (N + 255) / 256;
    const int eGrid    = (E + 255) / 256;
    const int e4Grid   = (4 * E + 255) / 256;

    // pre-convert all GEMM weights to f16 col-major fragment order
    cvt_colmajor_f16<<<16, 256, 0, stream>>>(W_in, w16_in, 64);
    for (int m = 0; m < 2 * GNT; ++m)
        cvt_colmajor_f16<<<16, 256, 0, stream>>>(hl_W + (size_t)m * 64 * 64,
                                                 w16_hl + (size_t)m * 64 * 64, 64);
    for (int l = 0; l < 2; ++l)
        cvt_colmajor_f16<<<16, 256, 0, stream>>>(lin_W + (size_t)l * 70 * 64,
                                                 w16_msg + (size_t)l * 64 * 64, 64);
    cvt_colmajor_f16<<<4, 256, 0, stream>>>(W_out, w16_out, 16);

    // h = relu(x @ W_in + b_in)
    gemm64_wmma<64, true, true><<<gemmGrid, 256, 0, stream>>>(x, w16_in, b_in, h, N);

    for (int l = 0; l < 2; ++l) {
        const float* hlb_l   = hl_b + (size_t)l * GNT * 64;
        const float* etemb_l = et_emb + (size_t)l * 4 * GETE;
        const float* eaW_l   = ea_W + (size_t)l * 2 * GEAE;
        const float* attW_l  = att_W + (size_t)l * 139 * 2;
        const float* linW2_l = lin_W + (size_t)l * 70 * 64 + 64 * 64; // eae rows 64..69

        // hw = HeteroLinear(h)
        hetero_gemm64<<<gemmGrid, 256, 0, stream>>>(
            h, w16_hl + (size_t)l * GNT * 64 * 64, hlb_l, node_type, hw, N);

        // node-space hoisted projections
        node_att_proj<<<nGrid, 256, 0, stream>>>(hw, attW_l, ad, as_, N);
        gemm64_wmma<64, false, false><<<gemmGrid, 256, 0, stream>>>(
            hw, w16_msg + (size_t)l * 64 * 64, nullptr, msgn, N);

        // zero accumulators / stats (h reused as acc)
        (void)hipMemsetAsync(h, 0, (size_t)N * 64 * sizeof(float), stream);
        (void)hipMemsetAsync(mk, 0, (size_t)N * 2 * sizeof(unsigned), stream);
        (void)hipMemsetAsync(ssum, 0, (size_t)N * 2 * sizeof(float), stream);

        // edge passes: logits+max, exp+sum, weighted scatter
        edge_logits<<<eGrid, 256, 0, stream>>>(edge_idx, edge_type, edge_attr, ad, as_,
                                               etemb_l, eaW_l, attW_l, alpha, mk, E);
        edge_exp<<<eGrid, 256, 0, stream>>>(edge_idx, mk, alpha, ssum, E);
        edge_scatter<<<e4Grid, 256, 0, stream>>>(edge_idx, edge_attr, eaW_l, linW2_l,
                                                 msgn, alpha, ssum, h, E);
    }

    // out = h @ W_out + b_out
    gemm64_wmma<16, false, true><<<gemmGrid, 256, 0, stream>>>(h, w16_out, b_out, out, N);
}
